// NemotronHMTP_49151605735734
// MI455X (gfx1250) — matrix operand order repaired and checked by reading, not compiled, not run on hardware
//
#include <hip/hip_runtime.h>

// ---------------------------------------------------------------------------
// DeepseekV3-style MoE layer for MI455X (gfx1250, wave32, WMMA).
// Sparse routed path: tokens grouped per expert. Block tile 128x128 so an
// average expert (~96 tokens) needs a single M-tile -> each expert's 16 MB
// fp32 weight panel streams from HBM ~once (~1 GB total ~ 43-50 us floor at
// 23.3 TB/s). fp32 -> bf16 conversion happens in-register while staging to
// LDS; bf16 operands are staged with gfx1250 async HBM->LDS copies
// (global_load_async_to_lds_b128 + s_wait_asynccnt). Math runs on
// v_wmma_f32_16x16x32_bf16 with fp32 accumulation. No atomics anywhere:
// fully deterministic output. Workspace: ~72 MB of d_ws assumed.
// ---------------------------------------------------------------------------

typedef unsigned short u16;
typedef unsigned int   u32;
typedef unsigned long long u64;
typedef __attribute__((ext_vector_type(16))) __bf16 v16bf;
typedef __attribute__((ext_vector_type(8)))  float  v8f;

#define T_TOK   1024          // 2 * 512 tokens
#define HID     2048
#define N_EXP   64
#define TOPK    6
#define MOE_I   1024
#define SH_I    4096
#define PAIRS   (T_TOK * TOPK)

// ---- workspace layout (bytes) ---------------------------------------------
static constexpr size_t OFF_EIDX = 0;                                 // int [PAIRS]
static constexpr size_t OFF_TW   = OFF_EIDX + (size_t)PAIRS * 4;      // float[PAIRS]
static constexpr size_t OFF_PTOK = OFF_TW   + (size_t)PAIRS * 4;      // int [PAIRS]
static constexpr size_t OFF_PW   = OFF_PTOK + (size_t)PAIRS * 4;      // float[PAIRS]
static constexpr size_t OFF_PPOS = OFF_PW   + (size_t)PAIRS * 4;      // int [PAIRS]
static constexpr size_t OFF_EOFF = OFF_PPOS + (size_t)PAIRS * 4;      // int [64]
static constexpr size_t OFF_ECNT = OFF_EOFF + 256;                    // int [64]
static constexpr size_t OFF_MID  = ((OFF_ECNT + 256 + 255) / 256) * 256;       // bf16[T][SH_I]
static constexpr size_t OFF_HC   = OFF_MID + (size_t)T_TOK * SH_I * 2;         // bf16[PAIRS][MOE_I]
static constexpr size_t OFF_OUTP = OFF_HC  + (size_t)PAIRS * MOE_I * 2;        // f32 [PAIRS][HID]

// ---- helpers ---------------------------------------------------------------
__device__ inline u16 f2bf(float f) {               // fp32 -> bf16 RNE
  u32 u = __float_as_uint(f);
  u32 r = u + 0x7FFFu + ((u >> 16) & 1u);
  return (u16)(r >> 16);
}

union FragBF { v16bf v; u32 u[8]; };

// A fragment 16x32 bf16 (ISA 7.12.2): lanes 0-15 M=lane; VGPR0-3 hold K 0-7
// (lanes 0-15) / K 8-15 (lanes 16-31); VGPR4-7 hold K 16-23 / 24-31.
__device__ inline v16bf load_frag_a(const u16* lds_a, int mBase, int lane) {
  const int m  = mBase + (lane & 15);
  const int h8 = (lane >> 4) << 3;
  FragBF f;
#pragma unroll
  for (int p = 0; p < 8; ++p) {
    const int k = (p < 4) ? (2 * p + h8) : (16 + 2 * (p - 4) + h8);
    f.u[p] = *(const u32*)(lds_a + m * 32 + k);
  }
  return f.v;
}

// B fragment 32x16 bf16: lane n = lane&15; same K order as A: VGPR0-3 hold K 0-7
// (lanes 0-15) / K 8-15 (lanes 16-31); VGPR4-7 hold K 16-23 / 24-31. LDS stores B n-major ([n][k]) so
// each packed K-pair is one aligned 32-bit LDS read.
__device__ inline v16bf load_frag_b(const u16* lds_b, int nBase, int lane) {
  const int n   = nBase + (lane & 15);
  const int h16 = (lane >> 4) << 3;
  FragBF f;
#pragma unroll
  for (int p = 0; p < 8; ++p) {
    const int k = (p < 4) ? (2 * p + h16) : (16 + 2 * (p - 4) + h16);
    f.u[p] = *(const u32*)(lds_b + n * 32 + k);
  }
  return f.v;
}

// ---- LDS staging (256 threads; A tile 128x32, B tile 32->[128][32]) --------
__device__ inline void stage_a_f32(u16* lds_a, const float* A, int lda,
                                   const int* s_rows, int k0, int tid) {
  const int idx0 = tid << 4;          // 16 elements / thread
  const int m = idx0 >> 5;            // tid / 2
  const int k = idx0 & 31;            // 0 or 16
  const float* src = A + (size_t)s_rows[m] * lda + k0 + k;
  u16* d = lds_a + m * 32 + k;
#pragma unroll
  for (int i = 0; i < 4; ++i) {
    float4 f = *(const float4*)(src + (i << 2));
    d[i * 4 + 0] = f2bf(f.x);
    d[i * 4 + 1] = f2bf(f.y);
    d[i * 4 + 2] = f2bf(f.z);
    d[i * 4 + 3] = f2bf(f.w);
  }
}

// bf16 A tile: direct HBM -> LDS via gfx1250 async copy (no VGPR bounce).
// 128x32 bf16 = 8 KB = 256 threads x 2 x 16B. LDS byte address is the low
// 32 bits of the generic pointer (ISA aperture rule).
__device__ inline void stage_a_bf16_async(u16* lds_a, const u16* A, int lda,
                                          const int* s_rows, int k0, int tid) {
#pragma unroll
  for (int i = 0; i < 2; ++i) {
    const int idx0 = (tid + (i << 8)) << 3;   // 8 ushorts per op
    const int m = idx0 >> 5;
    const int k = idx0 & 31;
    const u64 src = (u64)(uintptr_t)(A + (size_t)s_rows[m] * lda + k0 + k);
    const u32 dst = (u32)(uintptr_t)(lds_a + m * 32 + k);
    asm volatile("global_load_async_to_lds_b128 %0, %1, off"
                 :: "v"(dst), "v"(src) : "memory");
  }
}

__device__ inline void stage_b_f32(u16* lds_b, const float* B, int ldb,
                                   int nTile, int k0, int tid) {
  const int k  = tid >> 3;            // 0..31
  const int n0 = (tid & 7) << 4;      // 0..112
  const float* src = B + (size_t)(k0 + k) * ldb + nTile + n0;
#pragma unroll
  for (int i = 0; i < 4; ++i) {
    float4 f = *(const float4*)(src + (i << 2));
    u16* d = lds_b + (n0 + (i << 2)) * 32 + k;          // transpose to [n][k]
    d[0]  = f2bf(f.x);
    d[32] = f2bf(f.y);
    d[64] = f2bf(f.z);
    d[96] = f2bf(f.w);
  }
}

// ---- core: 128(M) x 128(N) block tile, 8 waves, each wave 32x64 ------------
template <bool ABF16>
__device__ inline void gemm_tile(const void* A, int lda, const float* B, int ldb,
                                 int nTile, int K, const int* s_rows,
                                 u16* lds_a, u16* lds_b, v8f acc[2][4]) {
  const int tid  = threadIdx.x;
  const int lane = tid & 31;
  const int wv   = tid >> 5;
  const int mw   = (wv & 3) << 5;     // 0,32,64,96
  const int nw   = (wv >> 2) << 6;    // 0,64
#pragma unroll
  for (int mi = 0; mi < 2; ++mi)
#pragma unroll
    for (int j = 0; j < 4; ++j)
#pragma unroll
      for (int r = 0; r < 8; ++r) acc[mi][j][r] = 0.0f;

  for (int k0 = 0; k0 < K; k0 += 32) {
    __syncthreads();
    if (ABF16) stage_a_bf16_async(lds_a, (const u16*)A, lda, s_rows, k0, tid);
    else       stage_a_f32(lds_a, (const float*)A, lda, s_rows, k0, tid);
    stage_b_f32(lds_b, B, ldb, nTile, k0, tid);
    if (ABF16) asm volatile("s_wait_asynccnt 0x0" ::: "memory");
    __syncthreads();
    v16bf a0 = load_frag_a(lds_a, mw, lane);
    v16bf a1 = load_frag_a(lds_a, mw + 16, lane);
#pragma unroll
    for (int j = 0; j < 4; ++j) {
      v16bf bf = load_frag_b(lds_b, nw + (j << 4), lane);
      acc[0][j] = __builtin_amdgcn_wmma_f32_16x16x32_bf16(
          false, a0, false, bf, (short)0, acc[0][j], false, false);
      acc[1][j] = __builtin_amdgcn_wmma_f32_16x16x32_bf16(
          false, a1, false, bf, (short)0, acc[1][j], false, false);
    }
  }
}

// ---- kernel 1: gate + noaux_tc routing -------------------------------------
__global__ __launch_bounds__(64) void k_gate(const float* __restrict__ x,
                                             const float* __restrict__ gw,
                                             const float* __restrict__ gb,
                                             int* __restrict__ tok_eidx,
                                             float* __restrict__ tok_w) {
  __shared__ float xs[HID];
  __shared__ float sc[N_EXP];
  __shared__ float sb[N_EXP];
  const int t   = blockIdx.x;
  const int tid = threadIdx.x;        // 64 threads, one expert each
  const float* xr = x + (size_t)t * HID;
  for (int i = tid * 4; i < HID; i += 64 * 4) {
    float4 v = *(const float4*)(xr + i);
    xs[i] = v.x; xs[i + 1] = v.y; xs[i + 2] = v.z; xs[i + 3] = v.w;
  }
  __syncthreads();
  float a = 0.0f;
  const float* g = gw + (size_t)tid * HID;
  for (int i = 0; i < HID; i += 4) {
    float4 v = *(const float4*)(g + i);
    a = fmaf(xs[i], v.x, fmaf(xs[i+1], v.y, fmaf(xs[i+2], v.z, fmaf(xs[i+3], v.w, a))));
  }
  float s = 1.0f / (1.0f + expf(-a));
  sc[tid] = s;
  sb[tid] = s + gb[tid];
  __syncthreads();
  if (tid == 0) {
    float gs[8];
    for (int gi = 0; gi < 8; ++gi) {                     // top-2 sum per group
      float m1 = -1e30f, m2 = -1e30f;
      for (int i = 0; i < 8; ++i) {
        float v = sb[gi * 8 + i];
        if (v > m1) { m2 = m1; m1 = v; } else if (v > m2) { m2 = v; }
      }
      gs[gi] = m1 + m2;
    }
    bool gsel[8] = {false, false, false, false, false, false, false, false};
    for (int r = 0; r < 4; ++r) {                        // top-4 groups
      int bi = 0; float bv = -1e30f;
      for (int gi = 0; gi < 8; ++gi)
        if (!gsel[gi] && gs[gi] > bv) { bv = gs[gi]; bi = gi; }
      gsel[bi] = true;
    }
    bool esel[N_EXP];
    for (int e = 0; e < N_EXP; ++e) esel[e] = false;
    int eidx[TOPK]; float w[TOPK]; float sum = 0.0f;
    for (int r = 0; r < TOPK; ++r) {                     // top-6 experts
      int bi = 0; float bv = -1e30f;
      for (int e = 0; e < N_EXP; ++e) {
        if (!gsel[e >> 3] || esel[e]) continue;
        if (sb[e] > bv) { bv = sb[e]; bi = e; }
      }
      esel[bi] = true; eidx[r] = bi; w[r] = sc[bi]; sum += sc[bi];
    }
    const float inv = 2.5f / (sum + 1e-20f);
    for (int r = 0; r < TOPK; ++r) {
      tok_eidx[t * TOPK + r] = eidx[r];
      tok_w[t * TOPK + r]    = w[r] * inv;
    }
  }
}

// ---- kernel 2: deterministic per-expert compaction (counting sort) ---------
__global__ __launch_bounds__(64) void k_compact(const int* __restrict__ tok_eidx,
                                                const float* __restrict__ tok_w,
                                                int* __restrict__ pair_tok,
                                                float* __restrict__ pair_w,
                                                int* __restrict__ pair_pos,
                                                int* __restrict__ e_off,
                                                int* __restrict__ e_cnt) {
  __shared__ int cnts[N_EXP];
  __shared__ int offs[N_EXP];
  const int e = threadIdx.x;
  int c = 0;
  for (int i = 0; i < PAIRS; ++i)
    if (tok_eidx[i] == e) ++c;
  cnts[e] = c;
  __syncthreads();
  if (e == 0) {
    int s = 0;
    for (int k = 0; k < N_EXP; ++k) { offs[k] = s; s += cnts[k]; }
  }
  __syncthreads();
  int pos = offs[e];
  for (int i = 0; i < PAIRS; ++i) {
    if (tok_eidx[i] == e) {
      pair_tok[pos] = i / TOPK;
      pair_w[pos]   = tok_w[i];
      pair_pos[i]   = pos;
      ++pos;
    }
  }
  e_off[e] = offs[e];
  e_cnt[e] = cnts[e];
}

// ---- kernel 3: shared up: mid = relu2(x @ Wup)  [1024 x 4096] bf16 ---------
__global__ __launch_bounds__(256) void k_up_shared(const float* __restrict__ x,
                                                   const float* __restrict__ wup,
                                                   u16* __restrict__ mid) {
  __shared__ u16 lds_a[128 * 32];
  __shared__ u16 lds_b[128 * 32];
  __shared__ int s_rows[128];
  const int tid = threadIdx.x;
  const int mTile = blockIdx.y << 7;
  const int nTile = blockIdx.x << 7;
  if (tid < 128) s_rows[tid] = mTile + tid;
  v8f acc[2][4];
  gemm_tile<false>(x, HID, wup, SH_I, nTile, HID, s_rows, lds_a, lds_b, acc);
  const int lane = tid & 31, wv = tid >> 5;
  const int mw = (wv & 3) << 5, nw = (wv >> 2) << 6;
  const int hs = lane >> 4, nI = lane & 15;
#pragma unroll
  for (int mi = 0; mi < 2; ++mi)
#pragma unroll
    for (int j = 0; j < 4; ++j)
#pragma unroll
      for (int r = 0; r < 8; ++r) {
        const int m = mTile + mw + (mi << 4) + r + (hs << 3);
        const int n = nTile + nw + (j << 4) + nI;
        const float rl = fmaxf(acc[mi][j][r], 0.0f);
        mid[(size_t)m * SH_I + n] = f2bf(rl * rl);
      }
}

// ---- kernel 4: shared down: out = mid @ Wdown  [1024 x 2048] f32 -----------
__global__ __launch_bounds__(256) void k_down_shared(const u16* __restrict__ mid,
                                                     const float* __restrict__ wdn,
                                                     float* __restrict__ out) {
  __shared__ u16 lds_a[128 * 32];
  __shared__ u16 lds_b[128 * 32];
  __shared__ int s_rows[128];
  const int tid = threadIdx.x;
  const int mTile = blockIdx.y << 7;
  const int nTile = blockIdx.x << 7;
  if (tid < 128) s_rows[tid] = mTile + tid;
  v8f acc[2][4];
  gemm_tile<true>(mid, SH_I, wdn, HID, nTile, SH_I, s_rows, lds_a, lds_b, acc);
  const int lane = tid & 31, wv = tid >> 5;
  const int mw = (wv & 3) << 5, nw = (wv >> 2) << 6;
  const int hs = lane >> 4, nI = lane & 15;
#pragma unroll
  for (int mi = 0; mi < 2; ++mi)
#pragma unroll
    for (int j = 0; j < 4; ++j)
#pragma unroll
      for (int r = 0; r < 8; ++r) {
        const int m = mTile + mw + (mi << 4) + r + (hs << 3);
        const int n = nTile + nw + (j << 4) + nI;
        out[(size_t)m * HID + n] = acc[mi][j][r];
      }
}

// ---- kernel 5: expert up: Hc = relu2(gather(x) @ W1[e]) * w  (bf16) --------
__global__ __launch_bounds__(256) void k_exp1(const float* __restrict__ x,
                                              const float* __restrict__ w1,
                                              const int* __restrict__ e_off,
                                              const int* __restrict__ e_cnt,
                                              const int* __restrict__ pair_tok,
                                              const float* __restrict__ pair_w,
                                              u16* __restrict__ Hc) {
  __shared__ u16 lds_a[128 * 32];
  __shared__ u16 lds_b[128 * 32];
  __shared__ int s_rows[128];
  __shared__ float s_w[128];
  const int e   = blockIdx.z;
  const int off = e_off[e];
  const int cnt = e_cnt[e];
  const int mTile = blockIdx.y << 7;
  if (mTile >= cnt) return;                 // uniform per block
  const int nTile = blockIdx.x << 7;
  const int tid = threadIdx.x;
  if (tid < 128) {
    const int ml = mTile + tid;
    const int p  = off + ((ml < cnt) ? ml : 0);
    s_rows[tid]  = pair_tok[p];
    s_w[tid]     = (ml < cnt) ? pair_w[p] : 0.0f;
  }
  v8f acc[2][4];
  gemm_tile<false>(x, HID, w1 + (size_t)e * HID * MOE_I, MOE_I,
                   nTile, HID, s_rows, lds_a, lds_b, acc);
  const int lane = tid & 31, wv = tid >> 5;
  const int mw = (wv & 3) << 5, nw = (wv >> 2) << 6;
  const int hs = lane >> 4, nI = lane & 15;
#pragma unroll
  for (int mi = 0; mi < 2; ++mi)
#pragma unroll
    for (int j = 0; j < 4; ++j)
#pragma unroll
      for (int r = 0; r < 8; ++r) {
        const int ml = mw + (mi << 4) + r + (hs << 3);
        if (mTile + ml < cnt) {
          const int n = nTile + nw + (j << 4) + nI;
          const float rl = fmaxf(acc[mi][j][r], 0.0f);
          Hc[(size_t)(off + mTile + ml) * MOE_I + n] = f2bf(rl * rl * s_w[ml]);
        }
      }
}

// ---- kernel 6: expert down: outp_row = Hc_row @ W2[e]  (f32 per pair) ------
__global__ __launch_bounds__(256) void k_exp2(const u16* __restrict__ Hc,
                                              const float* __restrict__ w2,
                                              const int* __restrict__ e_off,
                                              const int* __restrict__ e_cnt,
                                              float* __restrict__ outp) {
  __shared__ u16 lds_a[128 * 32];
  __shared__ u16 lds_b[128 * 32];
  __shared__ int s_rows[128];
  const int e   = blockIdx.z;
  const int off = e_off[e];
  const int cnt = e_cnt[e];
  const int mTile = blockIdx.y << 7;
  if (mTile >= cnt) return;
  const int nTile = blockIdx.x << 7;
  const int tid = threadIdx.x;
  if (tid < 128) {
    const int ml = mTile + tid;
    s_rows[tid] = off + ((ml < cnt) ? ml : (cnt - 1));
  }
  v8f acc[2][4];
  gemm_tile<true>(Hc, MOE_I, w2 + (size_t)e * MOE_I * HID, HID,
                  nTile, MOE_I, s_rows, lds_a, lds_b, acc);
  const int lane = tid & 31, wv = tid >> 5;
  const int mw = (wv & 3) << 5, nw = (wv >> 2) << 6;
  const int hs = lane >> 4, nI = lane & 15;
#pragma unroll
  for (int mi = 0; mi < 2; ++mi)
#pragma unroll
    for (int j = 0; j < 4; ++j)
#pragma unroll
      for (int r = 0; r < 8; ++r) {
        const int ml = mw + (mi << 4) + r + (hs << 3);
        if (mTile + ml < cnt) {
          const int n = nTile + nw + (j << 4) + nI;
          outp[(size_t)(off + mTile + ml) * HID + n] = acc[mi][j][r];
        }
      }
}

// ---- kernel 7: out[t] = shared[t] + sum_j outp[pos(t,j)]  (deterministic) --
__global__ __launch_bounds__(256) void k_reduce(const float* __restrict__ outp,
                                                const int* __restrict__ pair_pos,
                                                float* __restrict__ out) {
  const int idx = blockIdx.x * 256 + threadIdx.x;
  const int t = idx >> 11;          // / HID
  const int h = idx & (HID - 1);
  float a = out[idx];
#pragma unroll
  for (int j = 0; j < TOPK; ++j) {
    const int pos = pair_pos[t * TOPK + j];
    a += outp[(size_t)pos * HID + h];
  }
  out[idx] = a;
}

// ---------------------------------------------------------------------------
extern "C" void kernel_launch(void* const* d_in, const int* in_sizes, int n_in,
                              void* d_out, int out_size, void* d_ws, size_t ws_size,
                              hipStream_t stream) {
  (void)in_sizes; (void)n_in; (void)out_size; (void)ws_size;
  const float* x   = (const float*)d_in[0];   // [2,512,2048] -> [1024,2048]
  const float* gw  = (const float*)d_in[1];   // [64,2048]
  const float* gb  = (const float*)d_in[2];   // [64]
  const float* wup = (const float*)d_in[3];   // [2048,4096]
  const float* wdn = (const float*)d_in[4];   // [4096,2048]
  const float* w1  = (const float*)d_in[5];   // [64,2048,1024]
  const float* w2  = (const float*)d_in[6];   // [64,1024,2048]
  float* out = (float*)d_out;

  char* ws = (char*)d_ws;
  int*   tok_eidx = (int*)  (ws + OFF_EIDX);
  float* tok_w    = (float*)(ws + OFF_TW);
  int*   pair_tok = (int*)  (ws + OFF_PTOK);
  float* pair_w   = (float*)(ws + OFF_PW);
  int*   pair_pos = (int*)  (ws + OFF_PPOS);
  int*   e_off    = (int*)  (ws + OFF_EOFF);
  int*   e_cnt    = (int*)  (ws + OFF_ECNT);
  u16*   mid      = (u16*)  (ws + OFF_MID);
  u16*   Hc       = (u16*)  (ws + OFF_HC);
  float* outp     = (float*)(ws + OFF_OUTP);

  k_gate<<<T_TOK, 64, 0, stream>>>(x, gw, gb, tok_eidx, tok_w);
  k_compact<<<1, 64, 0, stream>>>(tok_eidx, tok_w, pair_tok, pair_w, pair_pos,
                                  e_off, e_cnt);
  k_up_shared<<<dim3(SH_I / 128, T_TOK / 128), 256, 0, stream>>>(x, wup, mid);
  k_down_shared<<<dim3(HID / 128, T_TOK / 128), 256, 0, stream>>>(mid, wdn, out);
  k_exp1<<<dim3(MOE_I / 128, T_TOK / 128, N_EXP), 256, 0, stream>>>(
      x, w1, e_off, e_cnt, pair_tok, pair_w, Hc);
  k_exp2<<<dim3(HID / 128, T_TOK / 128, N_EXP), 256, 0, stream>>>(
      Hc, w2, e_off, e_cnt, outp);
  k_reduce<<<(T_TOK * HID) / 256, 256, 0, stream>>>(outp, pair_pos, out);
}
